// MambaBlock_68676527063207
// MI455X (gfx1250) — matrix operand order repaired and checked by reading, not compiled, not run on hardware
//
#include <hip/hip_runtime.h>
#include <math.h>

// ---------------- problem constants (match reference) ----------------
#define B_     2
#define N_     1024
#define C_     256
#define S_     2
#define H_     8
#define HD_    32
#define DI_    512      // d_inner = EXPAND*C
#define DEPTH_N 4
#define DTR_   16
#define DST_   16

#define LSTR   40       // bf16 LDS row stride (80 bytes, 16B aligned)

typedef __attribute__((ext_vector_type(16))) __bf16 vbf16;
typedef __attribute__((ext_vector_type(8)))  float  v8f;

__device__ __forceinline__ v8f vzero8() {
  v8f z = {0.f,0.f,0.f,0.f,0.f,0.f,0.f,0.f};
  return z;
}

__device__ __forceinline__ unsigned short bfbits(float f) {
  unsigned u = __builtin_bit_cast(unsigned, f);
  u += 0x7FFFu + ((u >> 16) & 1u);           // round-to-nearest-even
  return (unsigned short)(u >> 16);
}
__device__ __forceinline__ unsigned pack2bf(float lo, float hi) {
  return (unsigned)bfbits(lo) | ((unsigned)bfbits(hi) << 16);
}
__device__ __forceinline__ __bf16 f2bf(float f) {
  unsigned short h = bfbits(f);
  return __builtin_bit_cast(__bf16, h);
}

// ---- A fragment 16(M)x32(K) from bf16 LDS tile [16][LSTR] (packed b128 loads)
// ISA 7.12.2: lane(0-15) K{0..7,16..23}, lane(16-31) K{8..15,24..31}
__device__ __forceinline__ vbf16 frag_a_bf(const __bf16* base, int lane) {
  int m = lane & 15, half = lane >> 4;
  const __bf16* row = base + m * LSTR;
  vbf16 a;
  uint4* pa = reinterpret_cast<uint4*>(&a);
  pa[0] = *reinterpret_cast<const uint4*>(row + half * 8);
  pa[1] = *reinterpret_cast<const uint4*>(row + 16 + half * 8);
  return a;
}

// ---- B fragment 32(K)x16(N) from bf16 LDS tile [16 n-rows][LSTR]
// lanes 0-15 carry K{0..7,16..23}, lanes 16-31 carry K{8..15,24..31} for column n=lane&15
__device__ __forceinline__ vbf16 frag_b_bf(const __bf16* base, int lane) {
  int n = lane & 15, kh = lane >> 4;
  const __bf16* row = base + n * LSTR;
  vbf16 b;
  uint4* pb = reinterpret_cast<uint4*>(&b);
  pb[0] = *reinterpret_cast<const uint4*>(row + kh * 8);
  pb[1] = *reinterpret_cast<const uint4*>(row + 16 + kh * 8);
  return b;
}

// ---- A fragment from an f32 LDS region (used for attention probabilities)
__device__ __forceinline__ vbf16 frag_a_f32(const float* base, int ld, int lane) {
  int m = lane & 15, half = lane >> 4;
  const float* row = base + m * ld;
  vbf16 a;
#pragma unroll
  for (int e = 0; e < 16; ++e) {
    int v = e >> 1, p = e & 1;
    int k = (v < 4) ? (half * 8 + v * 2 + p) : (16 + half * 8 + (v - 4) * 2 + p);
    a[e] = f2bf(row[k]);
  }
  return a;
}

__device__ __forceinline__ v8f wmma_bf16(vbf16 a, vbf16 b, v8f c) {
  return __builtin_amdgcn_wmma_f32_16x16x32_bf16(false, a, false, b, (short)0, c,
                                                 false, false);
}

// activations: 0=none 1=gelu(erf) 2=silu 3=softplus
__device__ __forceinline__ float act_apply(float x, int act) {
  if (act == 1) return 0.5f * x * (1.0f + erff(x * 0.70710678118654752f));
  if (act == 2) return x / (1.0f + __expf(-x));
  if (act == 3) return (x > 20.f) ? x : log1pf(__expf(x));
  return x;
}

// ---------------- generic WMMA linear: Y = act(X*W^T + b) (+ R) ----------------
// 128 threads = 4 waves in 2x2; each wave owns a 32x32 region (2x2 WMMA tiles);
// block tile 64x64, K-chunks of 32, bf16 LDS staging with float4 fast path.
__global__ __launch_bounds__(128) void linear_wmma_kernel(
    const float* __restrict__ X, int ldx,
    const float* __restrict__ W, int ldw,
    const float* __restrict__ bias,
    const float* __restrict__ R, int ldr,
    float* __restrict__ Y, int ldy,
    int M, int K, int N, int act)
{
  __shared__ __align__(16) __bf16 sX[64 * LSTR];
  __shared__ __align__(16) __bf16 sW[64 * LSTR];
  const int bm = blockIdx.x * 64, bn = blockIdx.y * 64;
  const int tid = threadIdx.x, lane = tid & 31, wv = tid >> 5;
  const int wr = (wv >> 1) * 32, wc = (wv & 1) * 32;
  const bool aligned = ((ldx & 3) == 0) && ((ldw & 3) == 0) &&
                       (bm + 64 <= M) && (bn + 64 <= N);

  v8f acc[2][2];
  acc[0][0] = vzero8(); acc[0][1] = vzero8();
  acc[1][0] = vzero8(); acc[1][1] = vzero8();

  for (int k0 = 0; k0 < K; k0 += 32) {
    if (aligned && (k0 + 32 <= K)) {
      // fast path: one float4 per (row, 4-col group); 128 threads cover 64x32
      for (int i = tid; i < 64 * 8; i += 128) {
        int r = i >> 3, c4 = (i & 7) * 4;
        float4 xv = *reinterpret_cast<const float4*>(X + (size_t)(bm + r) * ldx + k0 + c4);
        uint2 px; px.x = pack2bf(xv.x, xv.y); px.y = pack2bf(xv.z, xv.w);
        *reinterpret_cast<uint2*>(sX + r * LSTR + c4) = px;
        float4 wv4 = *reinterpret_cast<const float4*>(W + (size_t)(bn + r) * ldw + k0 + c4);
        uint2 pw; pw.x = pack2bf(wv4.x, wv4.y); pw.y = pack2bf(wv4.z, wv4.w);
        *reinterpret_cast<uint2*>(sW + r * LSTR + c4) = pw;
      }
    } else {
      // slow path: per-element bounds-checked, zero padded
      for (int i = tid; i < 64 * 16; i += 128) {
        int r = i >> 4, c2 = (i & 15) * 2;
        int gk0 = k0 + c2, gk1 = k0 + c2 + 1;
        int gm = bm + r;
        float x0 = (gm < M && gk0 < K) ? X[(size_t)gm * ldx + gk0] : 0.f;
        float x1 = (gm < M && gk1 < K) ? X[(size_t)gm * ldx + gk1] : 0.f;
        *reinterpret_cast<unsigned*>(sX + r * LSTR + c2) = pack2bf(x0, x1);
        int gn = bn + r;
        float w0 = (gn < N && gk0 < K) ? W[(size_t)gn * ldw + gk0] : 0.f;
        float w1 = (gn < N && gk1 < K) ? W[(size_t)gn * ldw + gk1] : 0.f;
        *reinterpret_cast<unsigned*>(sW + r * LSTR + c2) = pack2bf(w0, w1);
      }
    }
    // speculative prefetch of next K-chunk (global_prefetch_b8)
    if (k0 + 32 < K) {
      int gm = bm + (tid & 63);
      if (gm < M) __builtin_prefetch(X + (size_t)gm * ldx + k0 + 32, 0, 1);
    }
    __syncthreads();

    vbf16 a0 = frag_a_bf(sX + (wr + 0)  * LSTR, lane);
    vbf16 a1 = frag_a_bf(sX + (wr + 16) * LSTR, lane);
    vbf16 b0 = frag_b_bf(sW + (wc + 0)  * LSTR, lane);
    vbf16 b1 = frag_b_bf(sW + (wc + 16) * LSTR, lane);
    acc[0][0] = wmma_bf16(a0, b0, acc[0][0]);
    acc[0][1] = wmma_bf16(a0, b1, acc[0][1]);
    acc[1][0] = wmma_bf16(a1, b0, acc[1][0]);
    acc[1][1] = wmma_bf16(a1, b1, acc[1][1]);
    __syncthreads();
  }

#pragma unroll
  for (int ti = 0; ti < 2; ++ti) {
#pragma unroll
    for (int tj = 0; tj < 2; ++tj) {
#pragma unroll
      for (int r = 0; r < 8; ++r) {
        int m = bm + wr + ti * 16 + r + ((lane >> 4) << 3);
        int n = bn + wc + tj * 16 + (lane & 15);
        if (m < M && n < N) {
          float v = acc[ti][tj][r];
          if (bias) v += bias[n];
          v = act_apply(v, act);
          if (R) v += R[(size_t)m * ldr + n];
          Y[(size_t)m * ldy + n] = v;
        }
      }
    }
  }
}

// ---------------- fused MVSA attention (per (b,h,16-query tile)) ----------------
// dynamic LDS layout (byte offsets, all 16B aligned):
static constexpr int ATTB_SC  = 0;                         // f32  [16][1024]
static constexpr int ATTB_QT  = ATTB_SC + 16 * 1024 * 4;   // bf16 [16][LSTR]
static constexpr int ATTB_KST = ATTB_QT + 16 * LSTR * 2;   // bf16 [4][16][LSTR]
static constexpr int ATTB_VST = ATTB_KST + 4 * 16 * LSTR * 2; // bf16 [4][32][LSTR]
static constexpr int ATTB_PO  = ATTB_VST + 4 * 32 * LSTR * 2; // f32 [4][16][32]
static constexpr int ATTB_ST  = ATTB_PO + 4 * 512 * 4;     // f32  [16][8]
static constexpr int ATTB_TOT = ATTB_ST + 16 * 8 * 4;

__global__ __launch_bounds__(128) void mvsa_attn_kernel(
    const float* __restrict__ q,    // (B,N,C)
    const float* __restrict__ kv,   // (B,N,2C) : k then v, head-major within C
    const float* __restrict__ a1p, const float* __restrict__ a2p,
    float* __restrict__ out)        // (B,N,C) pre-projection
{
  extern __shared__ char smem[];
  float*  sc  = reinterpret_cast<float*>(smem + ATTB_SC);
  __bf16* qtb = reinterpret_cast<__bf16*>(smem + ATTB_QT);
  __bf16* kst = reinterpret_cast<__bf16*>(smem + ATTB_KST);
  __bf16* vst = reinterpret_cast<__bf16*>(smem + ATTB_VST);
  float*  po  = reinterpret_cast<float*>(smem + ATTB_PO);
  float*  st  = reinterpret_cast<float*>(smem + ATTB_ST);

  const int q0 = blockIdx.x * 16;
  const int h  = blockIdx.y;
  const int b  = blockIdx.z;
  const int tid = threadIdx.x, lane = tid & 31, wv = tid >> 5;
  const float scale = 0.17677669529663687f;   // 1/sqrt(32)
  const float a1 = a1p[0], a2 = a2p[0];

  // stage Q tile 16x32 as bf16 (one float4 per thread; 128 threads = 16x8 groups)
  {
    int r = tid >> 3, c4 = (tid & 7) * 4;
    float4 qv = *reinterpret_cast<const float4*>(
        q + ((size_t)(b * N_) + q0 + r) * C_ + h * HD_ + c4);
    uint2 pq; pq.x = pack2bf(qv.x, qv.y); pq.y = pack2bf(qv.z, qv.w);
    *reinterpret_cast<uint2*>(qtb + r * LSTR + c4) = pq;
  }
  __syncthreads();
  vbf16 afr = frag_a_bf(qtb, lane);

  // scores = Q K^T * scale  (wave wv handles 16 key-tiles)
  __bf16* myk = kst + wv * (16 * LSTR);
  for (int kt = wv * 16; kt < wv * 16 + 16; ++kt) {
    int key0 = kt * 16;
#pragma unroll
    for (int i = lane; i < 16 * 8; i += 32) {
      int r = i >> 3, c4 = (i & 7) * 4;
      float4 kvv = *reinterpret_cast<const float4*>(
          kv + ((size_t)(b * N_) + key0 + r) * (2 * C_) + h * HD_ + c4);
      uint2 pk; pk.x = pack2bf(kvv.x, kvv.y); pk.y = pack2bf(kvv.z, kvv.w);
      *reinterpret_cast<uint2*>(myk + r * LSTR + c4) = pk;
    }
    vbf16 bfr = frag_b_bf(myk, lane);
    v8f accs = wmma_bf16(afr, bfr, vzero8());
#pragma unroll
    for (int r = 0; r < 8; ++r) {
      int m = r + ((lane >> 4) << 3);
      int n = lane & 15;
      sc[m * 1024 + key0 + n] = accs[r] * scale;
    }
  }
  __syncthreads();

  // per-row: max/min, bisected top-k thresholds (k=512, k=341), softmax sums
  for (int j = 0; j < 4; ++j) {
    int r = wv * 4 + j;
    const float* row = sc + r * 1024;
    float mx = -3.4e38f, mn = 3.4e38f;
#pragma unroll 4
    for (int c = lane; c < 1024; c += 32) {
      float v = row[c];
      mx = fmaxf(mx, v); mn = fminf(mn, v);
    }
#pragma unroll
    for (int o = 16; o; o >>= 1) {
      mx = fmaxf(mx, __shfl_xor(mx, o, 32));
      mn = fminf(mn, __shfl_xor(mn, o, 32));
    }
    float th[2]; int kk[2] = {512, 341};
#pragma unroll 1
    for (int t = 0; t < 2; ++t) {
      float lo = mn, hi = mx;
#pragma unroll 1
      for (int it = 0; it < 22; ++it) {
        float mid = 0.5f * (lo + hi);
        int cnt = 0;
#pragma unroll 4
        for (int c = lane; c < 1024; c += 32) cnt += (row[c] >= mid) ? 1 : 0;
#pragma unroll
        for (int o = 16; o; o >>= 1) cnt += __shfl_xor(cnt, o, 32);
        if (cnt >= kk[t]) lo = mid; else hi = mid;
      }
      th[t] = lo;
    }
    float s1 = 0.f, s2 = 0.f;
#pragma unroll 2
    for (int c = lane; c < 1024; c += 32) {
      float v = row[c];
      float e = __expf(v - mx);
      if (v >= th[0]) s1 += e;
      if (v >= th[1]) s2 += e;
    }
#pragma unroll
    for (int o = 16; o; o >>= 1) { s1 += __shfl_xor(s1, o, 32); s2 += __shfl_xor(s2, o, 32); }
    if (lane == 0) {
      st[r * 8 + 0] = mx;
      st[r * 8 + 1] = th[0];
      st[r * 8 + 2] = th[1];
      st[r * 8 + 3] = a1 / fmaxf(s1, 1e-20f);
      st[r * 8 + 4] = a2 / fmaxf(s2, 1e-20f);
    }
  }
  __syncthreads();

  // scores -> combined dual-softmax probabilities (in place)
#pragma unroll 2
  for (int i = tid; i < 16 * 1024; i += 128) {
    int r = i >> 10;
    float v = sc[i];
    float e = __expf(v - st[r * 8 + 0]);
    float p = 0.f;
    if (v >= st[r * 8 + 1]) p += e * st[r * 8 + 3];
    if (v >= st[r * 8 + 2]) p += e * st[r * 8 + 4];
    sc[i] = p;
  }
  __syncthreads();

  // out = P @ V  (each wave accumulates 8 key-chunks of 32)
  __bf16* myv = vst + wv * (32 * LSTR);
  v8f acc0 = vzero8(), acc1 = vzero8();
  for (int kc = wv * 8; kc < wv * 8 + 8; ++kc) {
    int key0 = kc * 32;
    // stage V^T chunk as bf16: [dim(32)][key(32)], packed over key pairs
#pragma unroll 4
    for (int i = lane; i < 32 * 16; i += 32) {
      int d = i >> 4, kp = (i & 15) * 2;
      size_t r0 = ((size_t)(b * N_) + key0 + kp)     * (2 * C_) + C_ + h * HD_ + d;
      size_t r1 = ((size_t)(b * N_) + key0 + kp + 1) * (2 * C_) + C_ + h * HD_ + d;
      *reinterpret_cast<unsigned*>(myv + d * LSTR + kp) = pack2bf(kv[r0], kv[r1]);
    }
    vbf16 ap = frag_a_f32(sc + key0, 1024, lane);
    vbf16 b0 = frag_b_bf(myv, lane);
    vbf16 b1 = frag_b_bf(myv + 16 * LSTR, lane);
    acc0 = wmma_bf16(ap, b0, acc0);
    acc1 = wmma_bf16(ap, b1, acc1);
  }
#pragma unroll
  for (int r = 0; r < 8; ++r) {
    int m = r + ((lane >> 4) << 3);
    int n = lane & 15;
    po[wv * 512 + m * 32 + n]      = acc0[r];
    po[wv * 512 + m * 32 + 16 + n] = acc1[r];
  }
  __syncthreads();
  for (int i = tid; i < 512; i += 128) {
    int m = i >> 5, d = i & 31;
    float s = po[i] + po[512 + i] + po[1024 + i] + po[1536 + i];
    out[((size_t)(b * N_) + q0 + m) * C_ + h * HD_ + d] = s;
  }
}

// ---------------- elementwise / reduction kernels ----------------
__global__ void transpose_in_kernel(const float* __restrict__ pts, float* __restrict__ xf) {
  int t = blockIdx.x * blockDim.x + threadIdx.x;
  if (t >= B_ * N_ * C_) return;
  int c = t % C_; int bn = t / C_; int n = bn % N_; int b = bn / N_;
  xf[t] = pts[((size_t)b * C_ + c) * N_ + n];
}

// y = avgpool3 + avgpool5 + avgpool7 along N (zero padded)
__global__ void pool_kernel(const float* __restrict__ x, float* __restrict__ y) {
  int t = blockIdx.x * blockDim.x + threadIdx.x;
  if (t >= B_ * N_ * C_) return;
  int c = t % C_; int bn = t / C_; int n = bn % N_; int b = bn / N_;
  float s3 = 0.f, s5 = 0.f, s7 = 0.f;
#pragma unroll
  for (int j = -3; j <= 3; ++j) {
    int nn = n + j;
    float v = (nn >= 0 && nn < N_) ? x[((size_t)b * N_ + nn) * C_ + c] : 0.f;
    if (j >= -1 && j <= 1) s3 += v;
    if (j >= -2 && j <= 2) s5 += v;
    s7 += v;
  }
  y[t] = s3 * (1.f / 3.f) + s5 * (1.f / 5.f) + s7 * (1.f / 7.f);
}

__global__ __launch_bounds__(256) void ln_kernel(float* __restrict__ y,
                                                 const float* __restrict__ g,
                                                 const float* __restrict__ bb) {
  __shared__ float red[256];
  int row = blockIdx.x, tid = threadIdx.x;
  float v = y[(size_t)row * C_ + tid];
  red[tid] = v; __syncthreads();
  for (int s = 128; s > 0; s >>= 1) { if (tid < s) red[tid] += red[tid + s]; __syncthreads(); }
  float mu = red[0] * (1.f / C_); __syncthreads();
  float d = v - mu;
  red[tid] = d * d; __syncthreads();
  for (int s = 128; s > 0; s >>= 1) { if (tid < s) red[tid] += red[tid + s]; __syncthreads(); }
  float var = red[0] * (1.f / C_);
  y[(size_t)row * C_ + tid] = d * rsqrtf(var + 1e-5f) * g[tid] + bb[tid];
}

__global__ __launch_bounds__(256) void rms_kernel(const float* __restrict__ x,
                                                  const float* __restrict__ wgt,
                                                  float* __restrict__ y) {
  __shared__ float red[256];
  int row = blockIdx.x, tid = threadIdx.x;
  float v = x[(size_t)row * C_ + tid];
  red[tid] = v * v; __syncthreads();
  for (int s = 128; s > 0; s >>= 1) { if (tid < s) red[tid] += red[tid + s]; __syncthreads(); }
  float ms = red[0] * (1.f / C_);
  y[(size_t)row * C_ + tid] = v * rsqrtf(ms + 1e-6f) * wgt[tid];
}

// causal depthwise conv (D_CONV=4) on xz[..., :DI_] + bias, SiLU
__global__ void conv_silu_kernel(const float* __restrict__ xz,
                                 const float* __restrict__ cw,
                                 const float* __restrict__ cb,
                                 float* __restrict__ xc) {
  int t = blockIdx.x * blockDim.x + threadIdx.x;
  if (t >= B_ * N_ * DI_) return;
  int d = t % DI_; int bn = t / DI_; int n = bn % N_; int b = bn / N_;
  float acc = cb[d];
#pragma unroll
  for (int j = 0; j < 4; ++j) {
    int l = n - 3 + j;
    if (l >= 0) acc += cw[d * 4 + j] * xz[((size_t)b * N_ + l) * (2 * DI_) + d];
  }
  xc[t] = acc / (1.f + __expf(-acc));
}

// selective scan: one lane per (b, channel), 16 states in registers, B/C via LDS
__global__ __launch_bounds__(256) void scan_kernel(
    const float* __restrict__ dt, const float* __restrict__ xc,
    const float* __restrict__ dbl, const float* __restrict__ xz,
    const float* __restrict__ A_log, const float* __restrict__ Dp,
    float* __restrict__ y)
{
  int b = blockIdx.x >> 1;
  int d = ((blockIdx.x & 1) << 8) + threadIdx.x;
  float a[DST_], hreg[DST_];
#pragma unroll
  for (int n = 0; n < DST_; ++n) { a[n] = -__expf(A_log[d * DST_ + n]); hreg[n] = 0.f; }
  float Dpd = Dp[d];
  __shared__ float sB[DST_], sC[DST_];
#pragma unroll 1
  for (int l = 0; l < N_; ++l) {
    size_t row = (size_t)b * N_ + l;
    if (threadIdx.x < 16) sB[threadIdx.x] = dbl[row * 48 + 16 + threadIdx.x];
    else if (threadIdx.x < 32) sC[threadIdx.x - 16] = dbl[row * 48 + 32 + (threadIdx.x - 16)];
    __syncthreads();
    float dtv = dt[row * DI_ + d];
    float xv  = xc[row * DI_ + d];
    float dx  = dtv * xv;
    float yv = 0.f;
#pragma unroll
    for (int n = 0; n < DST_; ++n) {
      hreg[n] = __expf(dtv * a[n]) * hreg[n] + dx * sB[n];
      yv += hreg[n] * sC[n];
    }
    float zv = xz[row * (2 * DI_) + DI_ + d];
    float g  = zv / (1.f + __expf(-zv));
    y[row * DI_ + d] = (yv + xv * Dpd) * g;
    __syncthreads();
  }
}

__global__ void gather_kernel(const float* __restrict__ xf,
                              const float* __restrict__ coords,
                              const int* __restrict__ idx, int s,
                              float* __restrict__ sp, float* __restrict__ scb) {
  int t = blockIdx.x * blockDim.x + threadIdx.x;
  if (t >= B_ * N_ * C_) return;
  int c = t % C_; int bn = t / C_; int n = bn % N_; int b = bn / N_;
  int src = idx[((size_t)b * S_ + s) * N_ + n];
  sp[t] = xf[((size_t)b * N_ + src) * C_ + c];
  if (c < 3) scb[(size_t)bn * 3 + c] = coords[((size_t)b * 3 + c) * N_ + src];
}

// inverse-permutation scatter: accum[b, idx[n], :] += w * xm[b, n, :]
__global__ void scatter_kernel(const float* __restrict__ xm,
                               const int* __restrict__ idx, int s, float wsc,
                               float* __restrict__ accum) {
  int t = blockIdx.x * blockDim.x + threadIdx.x;
  if (t >= B_ * N_ * C_) return;
  int c = t % C_; int bn = t / C_; int n = bn % N_; int b = bn / N_;
  int dst = idx[((size_t)b * S_ + s) * N_ + n];
  accum[((size_t)b * N_ + dst) * C_ + c] += wsc * xm[t];
}

__global__ void zero_kernel(float* __restrict__ p, int n) {
  int t = blockIdx.x * blockDim.x + threadIdx.x;
  if (t < n) p[t] = 0.f;
}

// out[b,c,n] = xflat + res + gamma*mvsa   (with transpose back to (B,C,N))
__global__ void final_kernel(const float* __restrict__ xf,
                             const float* __restrict__ res,
                             const float* __restrict__ mv,
                             const float* __restrict__ gamma,
                             float* __restrict__ out) {
  int t = blockIdx.x * blockDim.x + threadIdx.x;
  if (t >= B_ * N_ * C_) return;
  int c = t % C_; int bn = t / C_; int n = bn % N_; int b = bn / N_;
  float v = xf[t] + res[t] + gamma[0] * mv[t];
  out[((size_t)b * C_ + c) * N_ + n] = v;
}

// ---------------- host driver ----------------
extern "C" void kernel_launch(void* const* d_in, const int* in_sizes, int n_in,
                              void* d_out, int out_size, void* d_ws, size_t ws_size,
                              hipStream_t stream) {
  (void)in_sizes; (void)n_in; (void)out_size; (void)ws_size;

  const float* pts    = (const float*)d_in[0];
  const int*   sidx   = (const int*)d_in[1];
  const float* coords = (const float*)d_in[2];
  const float* q_w    = (const float*)d_in[4];
  const float* q_b    = (const float*)d_in[5];
  const float* kv_w   = (const float*)d_in[6];
  const float* kv_b   = (const float*)d_in[7];
  const float* mv_w   = (const float*)d_in[8];
  const float* mv_b   = (const float*)d_in[9];
  const float* ln_g   = (const float*)d_in[10];
  const float* ln_bp  = (const float*)d_in[11];
  const float* attn1  = (const float*)d_in[12];
  const float* attn2  = (const float*)d_in[13];
  const float* gamma  = (const float*)d_in[14];
  const float* pe_w1  = (const float*)d_in[15];
  const float* pe_b1  = (const float*)d_in[16];
  const float* pe_w2  = (const float*)d_in[17];
  const float* pe_b2  = (const float*)d_in[18];
  const float* norm_w = (const float*)d_in[19];
  const float* in_w   = (const float*)d_in[20];
  const float* conv_w = (const float*)d_in[21];
  const float* conv_b = (const float*)d_in[22];
  const float* xproj_w= (const float*)d_in[23];
  const float* dt_w   = (const float*)d_in[24];
  const float* dt_b   = (const float*)d_in[25];
  const float* A_log  = (const float*)d_in[26];
  const float* Dp     = (const float*)d_in[27];
  const float* out_w  = (const float*)d_in[28];
  const float* f_w1   = (const float*)d_in[29];
  const float* f_b1   = (const float*)d_in[30];
  const float* f_w2   = (const float*)d_in[31];
  const float* f_b2   = (const float*)d_in[32];
  const float* res_w  = (const float*)d_in[33];
  const float* res_b  = (const float*)d_in[34];
  float* outp = (float*)d_out;

  // ---- workspace carve ----
  float* w = (float*)d_ws;
  size_t off = 0;
  auto carve = [&](size_t n) { float* p = w + off; off += n; return p; };
  const size_t BNC = (size_t)B_ * N_ * C_;
  float* xflat  = carve(BNC);
  float* ypool  = carve(BNC);
  float* qbuf   = carve(BNC);
  float* kvbuf  = carve((size_t)B_ * N_ * 2 * C_);
  float* attno  = carve(BNC);
  float* mvsao  = carve(BNC);
  float* accum  = carve(BNC);
  float* sp     = carve(BNC);
  float* scb    = carve((size_t)B_ * N_ * 4);   // rows of 3 coords (stride 3); over-allocated
  float* pehid  = carve((size_t)B_ * N_ * (C_ / 2));
  float* xsA    = carve(BNC);
  float* xsB    = carve(BNC);
  float* xn     = carve(BNC);
  float* xz     = carve((size_t)B_ * N_ * 2 * DI_);
  float* xc     = carve((size_t)B_ * N_ * DI_);
  float* dbl    = carve((size_t)B_ * N_ * 48);
  float* dtb    = carve((size_t)B_ * N_ * DI_);
  float* yb     = carve((size_t)B_ * N_ * DI_);
  float* mo     = carve(BNC);
  float* ff1    = carve(BNC);
  float* resb   = carve(BNC);

  auto cdiv = [](int a, int b) { return (a + b - 1) / b; };
  const int M = B_ * N_;                 // 2048 token rows
  const int TOT = B_ * N_ * C_;

  auto lin = [&](const float* X, int ldx, const float* Wt, int ldw,
                 const float* bias, const float* R, int ldr,
                 float* Y, int ldy, int Mm, int K, int Nn, int act) {
    dim3 g(cdiv(Mm, 64), cdiv(Nn, 64));
    linear_wmma_kernel<<<g, 128, 0, stream>>>(X, ldx, Wt, ldw, bias, R, ldr,
                                              Y, ldy, Mm, K, Nn, act);
  };

  // ---------- MVSA branch ----------
  transpose_in_kernel<<<cdiv(TOT, 256), 256, 0, stream>>>(pts, xflat);
  pool_kernel<<<cdiv(TOT, 256), 256, 0, stream>>>(xflat, ypool);
  ln_kernel<<<M, 256, 0, stream>>>(ypool, ln_g, ln_bp);
  lin(xflat, C_, q_w, C_, q_b, nullptr, 0, qbuf, C_, M, C_, C_, 0);
  lin(ypool, C_, kv_w, C_, kv_b, nullptr, 0, kvbuf, 2 * C_, M, C_, 2 * C_, 0);
  {
    dim3 g(N_ / 16, H_, B_);
    size_t smem = (size_t)ATTB_TOT;
    mvsa_attn_kernel<<<g, 128, smem, stream>>>(qbuf, kvbuf, attn1, attn2, attno);
  }
  lin(attno, C_, mv_w, C_, mv_b, nullptr, 0, mvsao, C_, M, C_, C_, 0);

  // ---------- Mamba scans ----------
  zero_kernel<<<cdiv(TOT, 256), 256, 0, stream>>>(accum, TOT);
  for (int s = 0; s < S_; ++s) {
    gather_kernel<<<cdiv(TOT, 256), 256, 0, stream>>>(xflat, coords, sidx, s, sp, scb);
    // positional embedding MLP, then x_seq = sp + pos
    lin(scb, 3, pe_w1, 3, pe_b1, nullptr, 0, pehid, C_ / 2, M, 3, C_ / 2, 1);
    lin(pehid, C_ / 2, pe_w2, C_ / 2, pe_b2, sp, C_, xsA, C_, M, C_ / 2, C_, 0);

    float* cur = xsA;
    float* nxt = xsB;
    for (int i = 0; i < DEPTH_N; ++i) {
      const float* in_wi   = in_w    + (size_t)i * (2 * DI_) * C_;
      const float* conv_wi = conv_w  + (size_t)i * DI_ * 4;
      const float* conv_bi = conv_b  + (size_t)i * DI_;
      const float* xproj_i = xproj_w + (size_t)i * 48 * DI_;
      const float* dt_wi   = dt_w    + (size_t)i * DI_ * DTR_;
      const float* dt_bi   = dt_b    + (size_t)i * DI_;
      const float* A_i     = A_log   + (size_t)i * DI_ * DST_;
      const float* Dp_i    = Dp      + (size_t)i * DI_;
      const float* out_wi  = out_w   + (size_t)i * C_ * DI_;
      const float* fw1     = f_w1    + (size_t)i * C_ * C_;
      const float* fb1     = f_b1    + (size_t)i * C_;
      const float* fw2     = f_w2    + (size_t)i * C_ * C_;
      const float* fb2     = f_b2    + (size_t)i * C_;

      rms_kernel<<<M, 256, 0, stream>>>(cur, norm_w + (size_t)i * C_, xn);
      lin(xn, C_, in_wi, C_, nullptr, nullptr, 0, xz, 2 * DI_, M, C_, 2 * DI_, 0);
      conv_silu_kernel<<<cdiv(B_ * N_ * DI_, 256), 256, 0, stream>>>(xz, conv_wi, conv_bi, xc);
      lin(xc, DI_, xproj_i, DI_, nullptr, nullptr, 0, dbl, 48, M, DI_, 48, 0);
      lin(dbl, 48, dt_wi, DTR_, dt_bi, nullptr, 0, dtb, DI_, M, DTR_, DI_, 3);
      scan_kernel<<<B_ * 2, 256, 0, stream>>>(dtb, xc, dbl, xz, A_i, Dp_i, yb);
      lin(yb, DI_, out_wi, DI_, nullptr, nullptr, 0, mo, C_, M, DI_, C_, 0);
      lin(mo, C_, fw1, C_, fb1, nullptr, 0, ff1, C_, M, C_, C_, 1);
      lin(ff1, C_, fw2, C_, fb2, cur, C_, nxt, C_, M, C_, C_, 0);
      float* tmp = cur; cur = nxt; nxt = tmp;
    }
    scatter_kernel<<<cdiv(TOT, 256), 256, 0, stream>>>(cur, sidx, s, 1.0f / S_, accum);
  }

  // ---------- residual head + output ----------
  lin(accum, C_, res_w, C_, res_b, nullptr, 0, resb, C_, M, C_, C_, 1);
  final_kernel<<<cdiv(TOT, 256), 256, 0, stream>>>(xflat, resb, mvsao, gamma, outp);
}
